// MassivePool_55894704390434
// MI455X (gfx1250) — matrix operand order repaired and checked by reading, not compiled, not run on hardware
//
#include <hip/hip_runtime.h>

// ---------------------------------------------------------------------------
// MassivePool retrieval on MI455X (gfx1250, wave32, WMMA)
//   scores    : bf16 WMMA 16x16x32, f32 accum; bf16 keys (67MB) L2-resident
//   top-k     : keys split into 4 slices (512 WGs) for WGP-level phase overlap;
//               streaming threshold filter in LDS; partial top-32 merged after
//   softmax   : wave32 shuffle reduction over 32 values
//   gather    : coalesced pool row gather, weighted sum
//   out GEMM  : f32 WMMA 16x16x4 (full fp32 output path)
// Scheduling: sched_barrier pins batched VMEM loads ahead of WMMA groups so
// waits become partial (s_wait_loadcnt N) instead of per-WMMA full stalls.
// ---------------------------------------------------------------------------

typedef __attribute__((ext_vector_type(16))) __bf16 v16bf;
typedef __attribute__((ext_vector_type(8)))  float  v8f;
typedef __attribute__((ext_vector_type(2)))  float  v2f;

#if __has_builtin(__builtin_amdgcn_sched_barrier)
#define SCHED_BARRIER() __builtin_amdgcn_sched_barrier(0)
#else
#define SCHED_BARRIER()
#endif

#define POOL_SIZE 262144
#define POOL_DIM  256
#define RET_DIM   128
#define TOP_K     32
#define NROWS     2048        // 2 * 1024 query rows
#define ROWS_WG   16          // query rows per workgroup (one WMMA M tile)
#define CHUNK     256         // keys processed per iteration (8 waves x 2 tiles)
#define SLICES    4           // key-dimension split for occupancy
#define KEYS_SLICE (POOL_SIZE / SLICES)
#define NEG_BIG   (-3.402823466e+38f)

__device__ __forceinline__ unsigned short f2bf(float f) {
  unsigned u = __builtin_bit_cast(unsigned, f);
  u += 0x7FFFu + ((u >> 16) & 1u);   // round to nearest even
  return (unsigned short)(u >> 16);
}

// ---------------------------------------------------------------- K0: f32->bf16
__global__ void convert_bf16_kernel(const float* __restrict__ in,
                                    unsigned short* __restrict__ out, int n) {
  int i = blockIdx.x * blockDim.x + threadIdx.x;
  if (i < n) out[i] = f2bf(in[i]);
}

// ------------------------------------- K1: scores (WMMA) + per-slice top-k
__global__ __launch_bounds__(256)
void scores_topk_kernel(const unsigned short* __restrict__ qbf,  // [2048][128] bf16
                        const unsigned short* __restrict__ kbf,  // [262144][128] bf16
                        float* __restrict__ pv_g,   // [SLICES][2048][32]
                        int*   __restrict__ pi_g) { // [SLICES][2048][32]
  __shared__ float sc[ROWS_WG][CHUNK];          // 16 KB streamed score tile
  __shared__ float topv[ROWS_WG][TOP_K];
  __shared__ int   topi[ROWS_WG][TOP_K];
  __shared__ float stage_v[ROWS_WG][CHUNK];     // threshold survivors
  __shared__ int   stage_i[ROWS_WG][CHUNK];
  __shared__ int   stage_n[ROWS_WG];
  __shared__ float rowmin[ROWS_WG];

  const int tid  = threadIdx.x;
  const int lane = tid & 31;
  const int wave = tid >> 5;         // 8 waves
  const int halfu = lane >> 4;       // 0 = lanes 0-15, 1 = lanes 16-31
  const int l15  = lane & 15;
  const int slice = blockIdx.x % SLICES;
  const int row0  = (blockIdx.x / SLICES) * ROWS_WG;
  const int kslice0 = slice * KEYS_SLICE;
  const int ksliceE = kslice0 + KEYS_SLICE;

  for (int e = tid; e < ROWS_WG * TOP_K; e += blockDim.x) {
    topv[e >> 5][e & 31] = NEG_BIG;
    topi[e >> 5][e & 31] = 0;
  }
  if (tid < ROWS_WG) { rowmin[tid] = NEG_BIG; stage_n[tid] = 0; }
  __syncthreads();

  // Preload A fragments (16 query rows, K=128 -> 4 fragments of K=32).
  // A 16x32 layout (contiguous, matching B): lane<16 holds K 0..15, lane>=16
  // holds K 16..31; 2 bf16 per VGPR -> two 16B loads per fragment.
  v16bf afrag[4];
  {
    const unsigned short* qrow = qbf + (size_t)(row0 + l15) * RET_DIM;
    #pragma unroll
    for (int kc = 0; kc < 4; ++kc) {
      const int kb = kc * 32 + (halfu ? 16 : 0);
      union { v16bf v; uint4 q[2]; } u;
      u.q[0] = *(const uint4*)(qrow + kb);        // K = kb .. kb+7
      u.q[1] = *(const uint4*)(qrow + kb + 8);    // K = kb+8 .. kb+15
      afrag[kc] = u.v;
    }
  }

  for (int key0 = kslice0; key0 < ksliceE; key0 += CHUNK) {
    // ---- each wave computes a 16x32 slab: 2 N-tiles of 16 keys
    const int kb0 = key0 + wave * 32;
    const unsigned short* krow0 = kbf + (size_t)(kb0 + l15) * RET_DIM;
    const unsigned short* krow1 = kbf + (size_t)(kb0 + 16 + l15) * RET_DIM;

    // prefetch next chunk's key rows into near caches
    if (key0 + CHUNK < ksliceE) {
      __builtin_prefetch(krow0 + (size_t)CHUNK * RET_DIM, 0, 3);
      __builtin_prefetch(krow1 + (size_t)CHUNK * RET_DIM, 0, 3);
    }

    // Issue ALL B-fragment loads for this chunk (batched clause), then a
    // scheduling barrier so the 8 WMMAs follow with only partial waits.
    // B in contiguous k order (same as A): lane<16 holds K 0..15, lane>=16 holds K 16..31,
    // so row-major bf16 key rows load directly, no transpose needed.
    v16bf bfr[8];
    #pragma unroll
    for (int kc = 0; kc < 4; ++kc) {
      const int kstart = kc * 32 + (halfu ? 16 : 0);
      bfr[kc * 2]     = *(const v16bf*)(krow0 + kstart);
      bfr[kc * 2 + 1] = *(const v16bf*)(krow1 + kstart);
    }
    SCHED_BARRIER();   // keep all 16 loads ahead of the WMMA group
    v8f acc0 = {}; v8f acc1 = {};
    #pragma unroll
    for (int kc = 0; kc < 4; ++kc) {
      acc0 = __builtin_amdgcn_wmma_f32_16x16x32_bf16(false, afrag[kc], false,
                bfr[kc * 2],     (short)0, acc0, false, false);
      acc1 = __builtin_amdgcn_wmma_f32_16x16x32_bf16(false, afrag[kc], false,
                bfr[kc * 2 + 1], (short)0, acc1, false, false);
    }
    // C layout: lane n = l15 is column, VGPR j is row j (+8 for lanes 16-31)
    const int col0 = wave * 32 + l15;
    #pragma unroll
    for (int j = 0; j < 8; ++j) {
      const int m = j + (halfu ? 8 : 0);
      sc[m][col0]      = acc0[j];
      sc[m][col0 + 16] = acc1[j];
    }
    __syncthreads();

    // ---- threshold scan: 16 threads per row, interleaved (bank-conflict-free)
    {
      const int r  = tid >> 4;
      const int c0 = tid & 15;
      const float thr = rowmin[r];
      #pragma unroll 4
      for (int i = 0; i < CHUNK / 16; ++i) {
        const int c = c0 + i * 16;
        const float v = sc[r][c];
        if (v > thr) {
          const int p = atomicAdd(&stage_n[r], 1);   // LDS ds_add_rtn
          stage_v[r][p] = v;
          stage_i[r][p] = key0 + c;
        }
      }
    }
    __syncthreads();

    // ---- rare serial insertions, one thread per row
    if (tid < ROWS_WG) {
      const int r = tid;
      const int n = stage_n[r];
      for (int j = 0; j < n; ++j) {
        const float v = stage_v[r][j];
        int slot = 0; float mv = topv[r][0];
        #pragma unroll
        for (int k = 1; k < TOP_K; ++k) {
          const float t = topv[r][k];
          if (t < mv) { mv = t; slot = k; }
        }
        if (v > mv) { topv[r][slot] = v; topi[r][slot] = stage_i[r][j]; }
      }
      if (n > 0) {
        float mv = topv[r][0];
        #pragma unroll
        for (int k = 1; k < TOP_K; ++k) mv = fminf(mv, topv[r][k]);
        rowmin[r] = mv;
      }
      stage_n[r] = 0;
    }
    __syncthreads();
  }

  for (int e = tid; e < ROWS_WG * TOP_K; e += blockDim.x) {
    const int r = e >> 5, k = e & 31;
    const size_t o = ((size_t)slice * NROWS + row0 + r) * TOP_K + k;
    pv_g[o] = topv[r][k];
    pi_g[o] = topi[r][k];
  }
}

// -------------------------------- K1b: merge per-slice top-32 -> global top-32
__global__ __launch_bounds__(128)
void merge_topk_kernel(const float* __restrict__ pv, const int* __restrict__ pi,
                       float* __restrict__ topv_g, int* __restrict__ topi_g) {
  __shared__ float bv[128][TOP_K];
  __shared__ int   bi[128][TOP_K];
  const int t = threadIdx.x;
  const int row = blockIdx.x * 128 + t;
  #pragma unroll
  for (int k = 0; k < TOP_K; ++k) { bv[t][k] = NEG_BIG; bi[t][k] = 0; }
  for (int s = 0; s < SLICES; ++s) {
    const size_t base = ((size_t)s * NROWS + row) * TOP_K;
    for (int k = 0; k < TOP_K; ++k) {
      const float v = pv[base + k];
      int slot = 0; float mv = bv[t][0];
      #pragma unroll
      for (int j = 1; j < TOP_K; ++j) {
        const float x = bv[t][j];
        if (x < mv) { mv = x; slot = j; }
      }
      if (v > mv) { bv[t][slot] = v; bi[t][slot] = pi[base + k]; }
    }
  }
  const size_t ob = (size_t)row * TOP_K;
  #pragma unroll
  for (int k = 0; k < TOP_K; ++k) { topv_g[ob + k] = bv[t][k]; topi_g[ob + k] = bi[t][k]; }
}

// ------------------------------------------ K2: softmax(32) + weighted gather
__global__ __launch_bounds__(256)
void softmax_gather_kernel(const float* __restrict__ topv_g,
                           const int*   __restrict__ topi_g,
                           const float* __restrict__ pool,     // [262144][256]
                           float* __restrict__ agg) {          // [2048][256]
  __shared__ float wsh[TOP_K];
  __shared__ int   ish[TOP_K];
  const int row = blockIdx.x;
  const int tid = threadIdx.x;
  if (tid < TOP_K) {
    const float v = topv_g[(size_t)row * TOP_K + tid];
    float m = v;
    #pragma unroll
    for (int o = 16; o > 0; o >>= 1) m = fmaxf(m, __shfl_xor(m, o, 32));
    const float e = __expf(v - m);
    float s = e;
    #pragma unroll
    for (int o = 16; o > 0; o >>= 1) s += __shfl_xor(s, o, 32);
    wsh[tid] = e / s;
    ish[tid] = topi_g[(size_t)row * TOP_K + tid];
  }
  __syncthreads();
  float acc = 0.f;
  #pragma unroll 8
  for (int k = 0; k < TOP_K; ++k)
    acc += wsh[k] * pool[(size_t)ish[k] * POOL_DIM + tid];
  agg[(size_t)row * POOL_DIM + tid] = acc;
}

// --------------------------------------- K3: out = agg @ w_out.T  (f32 WMMA)
__global__ __launch_bounds__(256)
void out_gemm_kernel(const float* __restrict__ agg,     // [2048][256]
                     const float* __restrict__ w_out,   // [256][256] row=out dim
                     float* __restrict__ out) {         // [2048][256]
  const int lane = threadIdx.x & 31;
  const int wave = threadIdx.x >> 5;
  const int halfu = lane >> 4;
  const int l15  = lane & 15;
  const int tile = blockIdx.x * 8 + wave;   // 128 M-tiles x 16 N-tiles = 2048
  const int m0 = (tile >> 4) * 16;
  const int n0 = (tile & 15) * 16;
  // f32 A 16x4: lane<16 -> K {0,1}, lane>=16 -> K {2,3}; B mirrors with N=lane
  const float* arow = agg   + (size_t)(m0 + l15) * POOL_DIM;
  const float* brow = w_out + (size_t)(n0 + l15) * POOL_DIM;  // B[k][n]=w_out[n][k]
  const int koff = halfu ? 2 : 0;
  v8f acc = {};
  #pragma unroll
  for (int kb = 0; kb < POOL_DIM; kb += 32) {   // 8 WMMAs per K block
    v2f a[8], b[8];
    #pragma unroll
    for (int i = 0; i < 8; ++i) {
      a[i] = *(const v2f*)(arow + kb + i * 4 + koff);
      b[i] = *(const v2f*)(brow + kb + i * 4 + koff);
    }
    SCHED_BARRIER();   // batch the 16 loads ahead of the 8 WMMAs
    #pragma unroll
    for (int i = 0; i < 8; ++i)
      acc = __builtin_amdgcn_wmma_f32_16x16x4_f32(false, a[i], false, b[i],
                                                  (short)0, acc, false, false);
  }
  const int n = n0 + l15;
  #pragma unroll
  for (int j = 0; j < 8; ++j) {
    const int m = m0 + j + (halfu ? 8 : 0);
    out[(size_t)m * POOL_DIM + n] = acc[j];
  }
}

// ---------------------------------------------------------------------------
extern "C" void kernel_launch(void* const* d_in, const int* in_sizes, int n_in,
                              void* d_out, int out_size, void* d_ws, size_t ws_size,
                              hipStream_t stream) {
  (void)in_sizes; (void)n_in; (void)out_size; (void)ws_size;
  const float* query = (const float*)d_in[0];   // [2,1024,128]
  const float* pool  = (const float*)d_in[1];   // [262144,256]
  const float* keys  = (const float*)d_in[2];   // [262144,128]
  const float* w_out = (const float*)d_in[3];   // [256,256]
  float* out = (float*)d_out;                   // [2,1024,256]

  char* ws = (char*)d_ws;
  size_t off = 0;
  unsigned short* kbf = (unsigned short*)(ws + off); off += (size_t)POOL_SIZE * RET_DIM * 2;
  unsigned short* qbf = (unsigned short*)(ws + off); off += (size_t)NROWS * RET_DIM * 2;
  float* pv   = (float*)(ws + off); off += (size_t)SLICES * NROWS * TOP_K * 4;
  int*   pi   = (int*)(ws + off);   off += (size_t)SLICES * NROWS * TOP_K * 4;
  float* topv = (float*)(ws + off); off += (size_t)NROWS * TOP_K * 4;
  int*   topi = (int*)(ws + off);   off += (size_t)NROWS * TOP_K * 4;
  float* agg  = (float*)(ws + off); off += (size_t)NROWS * POOL_DIM * 4;

  const int nk = POOL_SIZE * RET_DIM;
  convert_bf16_kernel<<<(nk + 255) / 256, 256, 0, stream>>>(keys, kbf, nk);
  const int nq = NROWS * RET_DIM;
  convert_bf16_kernel<<<(nq + 255) / 256, 256, 0, stream>>>(query, qbf, nq);

  scores_topk_kernel<<<(NROWS / ROWS_WG) * SLICES, 256, 0, stream>>>(qbf, kbf, pv, pi);
  merge_topk_kernel<<<NROWS / 128, 128, 0, stream>>>(pv, pi, topv, topi);
  softmax_gather_kernel<<<NROWS, 256, 0, stream>>>(topv, topi, pool, agg);
  out_gemm_kernel<<<(NROWS / 16) * (POOL_DIM / 16) / 8, 256, 0, stream>>>(agg, w_out, out);
}
